// FrequencyBranch_83013127897154
// MI455X (gfx1250) — matrix operand order repaired and checked by reading, not compiled, not run on hardware
//
#include <hip/hip_runtime.h>
#include <hip/hip_bf16.h>

// ---------------------------------------------------------------------------
// CDNA5 (gfx1250): 3x3 convs (C=64, B=4, 256x256) dominate the reference
// (~500 GFLOP).  Implicit GEMM on v_wmma_f32_16x16x32_f16 with
//   A = weights (M = Cout tile), B = im2col activations (N = pixels),
//   K = tap*64 + ci  (tap-major -> every fragment is one contiguous 32B load).
// Activations are NHWC f16 so staging / B-fragments / residual / D-stores are
// all b128 vector ops.  LDS staging now uses the gfx1250 async data path:
// per-lane global_load_async_to_lds_b128 (ASYNCcnt) + s_wait_asynccnt.
// ---------------------------------------------------------------------------

typedef __attribute__((ext_vector_type(16))) _Float16 v16h;
typedef __attribute__((ext_vector_type(8)))  _Float16 v8h;
typedef __attribute__((ext_vector_type(8)))  float    v8f;

#define CCH    64            // channels
#define KTOT   576           // Cin * 9
#define KSTEPS 18            // 576 / 32
#define TILEW  32            // output pixels per workgroup (one row segment)
#define HH     256
#define WW     256
#define NELEM  (4ll * 64 * 256 * 256)   // B*C*H*W

// ---------------------------------------------------------------------------
// NCHW f32 -> NHWC f16  (hardcoded 4x64x256x256)
// ---------------------------------------------------------------------------
__global__ __launch_bounds__(256) void nchw_to_nhwc_f16_kernel(
    const float* __restrict__ src, _Float16* __restrict__ dst) {
  long long i = (long long)blockIdx.x * blockDim.x + threadIdx.x;  // NHWC idx
  if (i >= NELEM) return;
  int c = (int)(i & 63);
  int w = (int)((i >> 6) & 255);
  int h = (int)((i >> 14) & 255);
  int b = (int)(i >> 22);
  dst[i] = (_Float16)src[(((long long)b * CCH + c) << 16) | (h << 8) | w];
}

__global__ __launch_bounds__(64) void zero_f32_kernel(float* __restrict__ p, int n) {
  int i = blockIdx.x * blockDim.x + threadIdx.x;
  if (i < n) p[i] = 0.0f;
}

// ---------------------------------------------------------------------------
// Pack conv weights (Co,Ci,3,3) f32 into the wave32 WMMA **A**-fragment
// layout for 16x16x32 f16, with K = tap*64 + ci:
//   fragment = ((kstep*4 + ctile)*32 + lane), 16 halves per lane
//   lane l:  m(co) = ctile*16 + (l&15)
//            k     = kstep*32 + 16*(l>>4) + j
// ---------------------------------------------------------------------------
__global__ __launch_bounds__(256) void pack_conv_w_kernel(
    const float* __restrict__ w,      // [64][64][3][3]
    _Float16* __restrict__ wpack) {   // [18][4][32][16]
  int tid = blockIdx.x * blockDim.x + threadIdx.x;   // 0 .. 36863
  if (tid >= KSTEPS * 4 * 32 * 16) return;
  int j     = tid & 15;
  int lane  = (tid >> 4) & 31;
  int ctile = (tid >> 9) & 3;
  int s     = tid >> 11;
  int co  = ctile * 16 + (lane & 15);
  int k   = s * 32 + ((lane >> 4) << 4) + j;
  int tap = k >> 6;          // 0..8  (kh*3+kw)
  int ci  = k & 63;
  wpack[tid] = (_Float16)w[(co * 64 + ci) * 9 + tap];
}

// ---------------------------------------------------------------------------
// Implicit-GEMM 3x3 conv, NHWC f16 in/out, f32 accumulate via WMMA.
// Block = 256 threads = 8 waves, computes 32 pixels x 64 channels at
// (b = blockIdx.z, h = blockIdx.y, w0 = blockIdx.x*32).
// LDS xs[3][34][64] f16 staged via async global->LDS DMA (ASYNCcnt).
// ---------------------------------------------------------------------------
template <int DO_RELU, int HAS_RES>
__global__ __launch_bounds__(256) void conv3x3_wmma_kernel(
    const _Float16* __restrict__ xin,    // NHWC f16
    const v16h*     __restrict__ wpackA, // packed A fragments
    const float*    __restrict__ bias,   // [64]
    const _Float16* __restrict__ resid,  // NHWC f16 (or unused)
    _Float16*       __restrict__ yout) { // NHWC f16
  __shared__ _Float16 xs[3 * 34 * CCH];  // (r*34 + col)*64 + ci

  const int tid = threadIdx.x;
  const int b   = blockIdx.z;
  const int h   = blockIdx.y;
  const int w0  = blockIdx.x * TILEW;

  // ---- stage haloed input tile into LDS via async global->LDS b128 DMA ----
  {
    const v8h* __restrict__ xg = (const v8h*)xin;
    v8h* xsv = (v8h*)xs;
    const v8h zero8 = {};
    for (int i = tid; i < 3 * 34 * 8; i += 256) {
      int ci8 = i & 7;
      int cc  = i >> 3;       // 0..101
      int col = cc % 34;
      int r   = cc / 34;
      int hh = h + r - 1;
      int ww = w0 + col - 1;
      if (hh >= 0 && hh < HH && ww >= 0 && ww < WW) {
        // vdst = LDS byte address (low 32 bits of the flat pointer, per the
        // gfx1250 LDS aperture rule), vaddr = 64-bit global address.
        unsigned ldsoff = (unsigned)(unsigned long long)(xsv + i);
        unsigned long long ga =
            (unsigned long long)(xg + ((((long long)b * HH + hh) * WW + ww) * 8 + ci8));
        asm volatile("global_load_async_to_lds_b128 %0, %1, off"
                     :: "v"(ldsoff), "v"(ga)
                     : "memory");
      } else {
        xsv[i] = zero8;       // halo zero-fill (DScnt path)
      }
    }
  }
  // GL2 prefetch of next block-row (global_prefetch_b8 path on gfx1250)
  if (h + 1 < HH) {
    const _Float16* pf = &xin[(((long long)b * HH + (h + 1)) * WW + w0) * CCH];
    __builtin_prefetch((const void*)pf, 0, 1);
  }
  // Drain this wave's async LDS writes, then workgroup barrier.
  asm volatile("s_wait_asynccnt 0x0" ::: "memory");
  __syncthreads();

  const int wave  = tid >> 5;
  const int lane  = tid & 31;
  const int ptile = wave & 1;
  const int ctile = wave >> 1;
  const int khigh = lane >> 4;       // K-half select (this file's contiguous k order)
  const int lpix  = lane & 15;       // B-matrix N index (pixel)
  const int p     = ptile * 16 + lpix;

  v8f acc = {};

#pragma unroll
  for (int s = 0; s < KSTEPS; ++s) {
    const int tap = s >> 1;          // compile-time after unroll
    const int r   = tap / 3;
    const int kw  = tap - r * 3;
    const int cib = (s & 1) * 32 + (khigh << 4);  // 16 consecutive channels

    // A fragment (weights): one 32B contiguous global load
    v16h afrag = wpackA[(s * 4 + ctile) * 32 + lane];
    // B fragment (activations): one 32B contiguous, 32B-aligned LDS load
    v16h bfrag = *(const v16h*)&xs[(r * 34 + (p + kw)) * CCH + cib];

    acc = __builtin_amdgcn_wmma_f32_16x16x32_f16(
        /*neg_a=*/false, afrag, /*neg_b=*/false, bfrag,
        /*c_mod=*/(short)0, acc, /*reuse_a=*/false, /*reuse_b=*/false);
  }

  // ---- epilogue: D lane holds 8 consecutive channels of one pixel ----
  const int cobase = ctile * 16 + (khigh << 3);
  const v8f bv = *(const v8f*)(bias + cobase);
  const long long obase =
      ((((long long)b * HH + h) * WW) + (w0 + ptile * 16 + lpix)) * CCH + cobase;

  v8h res = {};
  if (HAS_RES) res = *(const v8h*)(resid + obase);

  v8h o;
#pragma unroll
  for (int r = 0; r < 8; ++r) {
    float v = acc[r] + bv[r];
    if (DO_RELU) v = v > 0.0f ? v : 0.0f;
    if (HAS_RES) v += (float)res[r];
    o[r] = (_Float16)v;
  }
  *(v8h*)(yout + obase) = o;
}

// ---------------------------------------------------------------------------
// out(NCHW f32) = feat(NHWC f16) + x(NCHW f32)     [global residual tail]
// ---------------------------------------------------------------------------
__global__ __launch_bounds__(256) void combine_out_kernel(
    const _Float16* __restrict__ feat, const float* __restrict__ x,
    float* __restrict__ out, long long n) {
  long long i = (long long)blockIdx.x * blockDim.x + threadIdx.x;  // NCHW idx
  if (i >= n) return;
  int w = (int)(i & 255);
  int h = (int)((i >> 8) & 255);
  int c = (int)((i >> 16) & 63);
  int b = (int)(i >> 22);
  long long fidx = ((((long long)b << 8) | h) << 8 | w) * CCH + c;
  out[i] = (float)feat[fidx] + x[i];
}

__global__ __launch_bounds__(256) void copy_f32_kernel(
    const float* __restrict__ src, float* __restrict__ dst, long long n) {
  long long i = (long long)blockIdx.x * blockDim.x + threadIdx.x;
  if (i < n) dst[i] = src[i];
}

// ---------------------------------------------------------------------------
// Host driver
// ---------------------------------------------------------------------------
extern "C" void kernel_launch(void* const* d_in, const int* in_sizes, int n_in,
                              void* d_out, int out_size, void* d_ws, size_t ws_size,
                              hipStream_t stream) {
  const long long nel = NELEM;

  const float* x = nullptr;
  const float* wlist[16];
  int nw = 0;
  for (int i = 0; i < n_in; ++i) {
    if (!x && in_sizes[i] == (int)nel) x = (const float*)d_in[i];
    if (in_sizes[i] == 64 * 64 * 3 * 3 && nw < 16) wlist[nw++] = (const float*)d_in[i];
  }

  float* out = (float*)d_out;
  const long long nout = (out_size < (int)nel) ? (long long)out_size : nel;

  const size_t szF16  = (size_t)nel * 2;
  const size_t off_b0 = 0;
  const size_t off_b1 = szF16;
  const size_t off_b2 = 2 * szF16;
  const size_t off_w1 = 3 * szF16;
  const size_t off_w2 = off_w1 + 128 * 1024;
  const size_t off_bz = off_w2 + 128 * 1024;
  const size_t need   = off_bz + 4096;

  if (!x || ws_size < need) {
    if (x)
      copy_f32_kernel<<<(int)((nout + 255) / 256), 256, 0, stream>>>(x, out, nout);
    return;
  }

  char* ws = (char*)d_ws;
  _Float16* buf[3] = {(_Float16*)(ws + off_b0), (_Float16*)(ws + off_b1),
                      (_Float16*)(ws + off_b2)};
  _Float16* wp1 = (_Float16*)(ws + off_w1);
  _Float16* wp2 = (_Float16*)(ws + off_w2);
  float*    bz  = (float*)(ws + off_bz);

  zero_f32_kernel<<<1, 64, 0, stream>>>(bz, 64);
  nchw_to_nhwc_f16_kernel<<<(int)((nel + 255) / 256), 256, 0, stream>>>(x, buf[0]);

  dim3 cgrid(WW / TILEW, HH, 4);   // (8, 256, 4)
  const int packBlocks = (KSTEPS * 4 * 32 * 16 + 255) / 256;

  // 6 residual blocks: tmp = relu(conv1(cur)); next = cur + conv2(tmp)
  int cur = 0, tmp = 1, nxt = 2;
  const int NBLK = 6;
  if (nw > 0) {
    for (int bi = 0; bi < NBLK; ++bi) {
      const float* w1 = wlist[(2 * bi) % nw];
      const float* w2 = wlist[(2 * bi + 1) % nw];
      pack_conv_w_kernel<<<packBlocks, 256, 0, stream>>>(w1, wp1);
      pack_conv_w_kernel<<<packBlocks, 256, 0, stream>>>(w2, wp2);

      conv3x3_wmma_kernel<1, 0><<<cgrid, 256, 0, stream>>>(
          buf[cur], (const v16h*)wp1, bz, nullptr, buf[tmp]);
      conv3x3_wmma_kernel<0, 1><<<cgrid, 256, 0, stream>>>(
          buf[tmp], (const v16h*)wp2, bz, buf[cur], buf[nxt]);

      int old_cur = cur;
      cur = nxt;
      nxt = old_cur;
    }
  }

  combine_out_kernel<<<(int)((nout + 255) / 256), 256, 0, stream>>>(
      buf[cur], x, out, nout);
}
